// Encoder2_15814069584107
// MI455X (gfx1250) — matrix operand order repaired and checked by reading, not compiled, not run on hardware
//
#include <hip/hip_runtime.h>

typedef __attribute__((ext_vector_type(16))) _Float16 v16h;
typedef __attribute__((ext_vector_type(8)))  float    v8f;

#define BM 128
#define BN 64
#define BK 32
#define LDS_STRIDE (BK + 8)   // halves; 80 B/row -> 4B aligned, bank-conflict padded

union Frag  { v16h h; unsigned int u[8]; };
union Pack2 { _Float16 h[2]; unsigned int u; };

// ---- load next tile's data into registers (global loads issued here) ----
// Non-trans A: each thread owns 8 K-pairs of the 128x32 tile -> packed b32 LDS stores.
// Trans A (miRNA only): m-fastest scalar path.
template<bool TRANSA, bool USE_RDIV>
__device__ __forceinline__ void load_regs(
    const float* __restrict__ A, const float* __restrict__ B,
    const float* __restrict__ rdeg,
    float (&ra)[16], float (&rb)[8],
    int m0, int n0, int k0, int M, int N, int K, int lda, int tid)
{
    const bool fullA = (m0 + BM <= M) && (k0 + BK <= K);
    if (!TRANSA) {
        #pragma unroll
        for (int i = 0; i < 8; ++i) {
            int idx = tid + 256 * i;               // 2048 K-pairs
            int mm = idx >> 4, kk = (idx & 15) * 2;
            int gm = m0 + mm, gk = k0 + kk;
            float v0 = 0.f, v1 = 0.f;
            if (fullA || (gm < M && gk + 1 < K)) {
                const float* p = A + (long)gm * lda + gk;
                v0 = p[0]; v1 = p[1];
            } else if (gm < M && gk < K) {
                v0 = A[(long)gm * lda + gk];
            }
            if (USE_RDIV) {
                int gmc = gm < M ? gm : (M - 1);
                float r = rdeg[gmc];               // precomputed 1/max(deg,1)
                v0 *= r; v1 *= r;
            }
            ra[2 * i] = v0; ra[2 * i + 1] = v1;
        }
    } else {
        #pragma unroll
        for (int i = 0; i < 16; ++i) {
            int idx = tid + 256 * i;               // 4096 elems, m fastest (contiguous)
            int mm = idx & 127, kk = idx >> 7;
            int gm = m0 + mm, gk = k0 + kk;
            float v = 0.f;
            if (fullA || (gm < M && gk < K)) v = A[(long)gk * lda + gm];
            if (USE_RDIV) { int gmc = gm < M ? gm : (M - 1); v *= rdeg[gmc]; }
            ra[i] = v;
        }
    }
    const bool fullB = (k0 + BK <= K);             // n0+BN <= N always (N % 64 == 0)
    #pragma unroll
    for (int i = 0; i < 4; ++i) {
        int idx = tid + 256 * i;                   // 1024 K-pairs of 32x64 tile
        int nn = idx & 63, kk = (idx >> 6) * 2;
        int gk = k0 + kk, gn = n0 + nn;
        float v0 = 0.f, v1 = 0.f;
        if (fullB || gk + 1 < K) { v0 = B[(long)gk * N + gn]; v1 = B[(long)(gk + 1) * N + gn]; }
        else if (gk < K)         { v0 = B[(long)gk * N + gn]; }
        rb[2 * i] = v0; rb[2 * i + 1] = v1;
    }
}

// ---- convert (packed) + store staged registers into an LDS buffer ----
template<bool TRANSA>
__device__ __forceinline__ void store_lds(
    const float (&ra)[16], const float (&rb)[8],
    _Float16* __restrict__ Ah, _Float16* __restrict__ Bt, int tid)
{
    if (!TRANSA) {
        #pragma unroll
        for (int i = 0; i < 8; ++i) {
            int idx = tid + 256 * i;
            int mm = idx >> 4, kk = (idx & 15) * 2;
            Pack2 p; p.h[0] = (_Float16)ra[2 * i]; p.h[1] = (_Float16)ra[2 * i + 1];
            *(unsigned int*)(Ah + mm * LDS_STRIDE + kk) = p.u;   // ds_store_b32
        }
    } else {
        #pragma unroll
        for (int i = 0; i < 16; ++i) {
            int idx = tid + 256 * i;
            int mm = idx & 127, kk = idx >> 7;
            Ah[mm * LDS_STRIDE + kk] = (_Float16)ra[i];
        }
    }
    #pragma unroll
    for (int i = 0; i < 4; ++i) {
        int idx = tid + 256 * i;
        int nn = idx & 63, kk = (idx >> 6) * 2;
        Pack2 p; p.h[0] = (_Float16)rb[2 * i]; p.h[1] = (_Float16)rb[2 * i + 1];
        *(unsigned int*)(Bt + nn * LDS_STRIDE + kk) = p.u;       // Bt[n][k], ds_store_b32
    }
}

__device__ __forceinline__ void load_fragA(Frag& f, const _Float16* base, int row, int aseg) {
    const unsigned int* ap = (const unsigned int*)(base + row * LDS_STRIDE);
    #pragma unroll
    for (int j = 0; j < 4; ++j) f.u[j]     = ap[(aseg >> 1) + j];        // K=aseg+2j,+1
    #pragma unroll
    for (int j = 0; j < 4; ++j) f.u[4 + j] = ap[((16 + aseg) >> 1) + j]; // K=16+aseg+2j,+1
}
__device__ __forceinline__ void load_fragB(Frag& f, const _Float16* base, int col, int bseg) {
    const unsigned int* bp = (const unsigned int*)(base + col * LDS_STRIDE);
    #pragma unroll
    for (int j = 0; j < 4; ++j) f.u[j] = bp[(bseg >> 1) + j];            // K=bseg+2j,+1
    #pragma unroll
    for (int j = 0; j < 4; ++j) f.u[4 + j] = bp[((16 + bseg) >> 1) + j]; // K=16+bseg+2j,+1
}

// out[M x N] (+)= alpha * ( A(^T?, row-scaled?) @ B + bias ), optional accumulate + relu
template<bool TRANSA, bool USE_RDIV>
__global__ __launch_bounds__(256)
void gemm_wmma_f16(const float* __restrict__ A, const float* __restrict__ B,
                   const float* __restrict__ bias, const float* __restrict__ rdeg,
                   float* __restrict__ out,
                   int M, int N, int K, int lda,
                   float alpha, int acc, int relu)
{
    __shared__ _Float16 Ah[2][BM * LDS_STRIDE];
    __shared__ _Float16 Bt[2][BN * LDS_STRIDE];

    const int tid  = threadIdx.x;
    const int lane = tid & 31;
    const int wid  = tid >> 5;        // 8 waves: 4x2 grid of 32x32 wave tiles
    const int wr   = wid >> 1;
    const int wc   = wid & 1;
    const int m0   = blockIdx.x * BM;
    const int n0   = blockIdx.y * BN;

    v8f c00 = {}, c01 = {}, c10 = {}, c11 = {};
    float ra[16], rb[8];

    load_regs<TRANSA, USE_RDIV>(A, B, rdeg, ra, rb, m0, n0, 0, M, N, K, lda, tid);
    store_lds<TRANSA>(ra, rb, Ah[0], Bt[0], tid);
    __syncthreads();

    const int nk = (K + BK - 1) / BK;
    for (int kt = 0; kt < nk; ++kt) {
        const int cur = kt & 1;
        const bool more = (kt + 1) < nk;
        if (more)   // issue next tile's global loads; they fly while WMMA runs below
            load_regs<TRANSA, USE_RDIV>(A, B, rdeg, ra, rb, m0, n0, (kt + 1) * BK, M, N, K, lda, tid);

        // ---- fragments per CDNA5 wave32 16-bit layouts ----
        // A 16x32: lane<16 -> M=lane, K pairs {0..7,16..23}; lane>=16 -> {8..15,24..31}
        // B 32x16: lane<16 -> N=lane, K {0..7,16..23}; lane>=16 -> {8..15,24..31}
        const int arow = 32 * wr + (lane & 15);
        const int aseg = (lane & 16) ? 8 : 0;
        const int bcol = 32 * wc + (lane & 15);
        const int bseg = (lane & 16) ? 8 : 0;
        Frag fa0, fa1, fb0, fb1;
        load_fragA(fa0, Ah[cur], arow,      aseg);
        load_fragA(fa1, Ah[cur], arow + 16, aseg);
        load_fragB(fb0, Bt[cur], bcol,      bseg);
        load_fragB(fb1, Bt[cur], bcol + 16, bseg);

        c00 = __builtin_amdgcn_wmma_f32_16x16x32_f16(false, fa0.h, false, fb0.h, (short)0, c00, false, false);
        c01 = __builtin_amdgcn_wmma_f32_16x16x32_f16(false, fa0.h, false, fb1.h, (short)0, c01, false, false);
        c10 = __builtin_amdgcn_wmma_f32_16x16x32_f16(false, fa1.h, false, fb0.h, (short)0, c10, false, false);
        c11 = __builtin_amdgcn_wmma_f32_16x16x32_f16(false, fa1.h, false, fb1.h, (short)0, c11, false, false);

        __syncthreads();                       // LDS[cur] reads done before it is restaged
        if (more) {
            store_lds<TRANSA>(ra, rb, Ah[cur ^ 1], Bt[cur ^ 1], tid);
            __syncthreads();                   // next buffer ready before next compute
        }
    }

    // ---- epilogue: C/D layout: VGPR r -> M = r (+8 for lanes 16..31), N = lane&15 ----
    const int rbase = (lane & 16) ? 8 : 0;
    #pragma unroll
    for (int af = 0; af < 2; ++af) {
        #pragma unroll
        for (int r = 0; r < 8; ++r) {
            int row = m0 + 32 * wr + 16 * af + rbase + r;
            if (row >= M) continue;
            #pragma unroll
            for (int bf = 0; bf < 2; ++bf) {
                int col = n0 + 32 * wc + 16 * bf + (lane & 15);
                float v = (af == 0) ? ((bf == 0) ? c00[r] : c01[r])
                                    : ((bf == 0) ? c10[r] : c11[r]);
                if (bias) v += bias[col];
                v *= alpha;
                long o = (long)row * N + col;
                if (acc) v += out[o];
                if (relu) v = v > 0.f ? v : 0.f;
                out[o] = v;
            }
        }
    }
}

__global__ void fill_zero(float* __restrict__ p, long n) {
    long i = (long)blockIdx.x * blockDim.x + threadIdx.x;
    if (i < n) p[i] = 0.f;
}

__global__ void deg_kernel(const int* __restrict__ dst, float* __restrict__ deg, int E) {
    int e = blockIdx.x * blockDim.x + threadIdx.x;
    if (e < E) unsafeAtomicAdd(&deg[dst[e]], 1.0f);
}

// deg[i] := 1 / max(deg[i], 1)  (so GEMM staging multiplies instead of divides)
__global__ void finalize_rdeg(float* __restrict__ deg, int n) {
    int i = blockIdx.x * blockDim.x + threadIdx.x;
    if (i < n) { float d = deg[i]; deg[i] = 1.0f / (d > 1.f ? d : 1.f); }
}

// agg[dst] += h[src], 4 floats per thread; atomics resolve in L2 (agg fits easily)
__global__ void scatter_kernel(const float* __restrict__ h, const int* __restrict__ src,
                               const int* __restrict__ dst, float* __restrict__ agg,
                               int E, int D) {
    int idx = blockIdx.x * blockDim.x + threadIdx.x;
    int per = D >> 2;
    int e = idx / per;
    if (e >= E) return;
    int c = (idx - e * per) << 2;
    int s = src[e], d = dst[e];
    const float4 v = *(const float4*)(h + (long)s * D + c);
    float* o = agg + (long)d * D + c;
    unsafeAtomicAdd(o + 0, v.x);
    unsafeAtomicAdd(o + 1, v.y);
    unsafeAtomicAdd(o + 2, v.z);
    unsafeAtomicAdd(o + 3, v.w);
}

extern "C" void kernel_launch(void* const* d_in, const int* in_sizes, int n_in,
                              void* d_out, int out_size, void* d_ws, size_t ws_size,
                              hipStream_t stream) {
    const float* d_features = (const float*)d_in[0];
    const float* g_features = (const float*)d_in[1];
    const float* miRNA_dis  = (const float*)d_in[2];
    const float* gene_dis   = (const float*)d_in[3];
    const float* Wd  = (const float*)d_in[4];
    const float* bd  = (const float*)d_in[5];
    const float* Wg  = (const float*)d_in[6];
    const float* bg  = (const float*)d_in[7];
    const float* W1  = (const float*)d_in[8];
    const float* W4  = (const float*)d_in[9];
    const float* Ws1 = (const float*)d_in[10];
    const float* Wn1 = (const float*)d_in[11];
    const float* b1  = (const float*)d_in[12];
    const float* Ws2 = (const float*)d_in[13];
    const float* Wn2 = (const float*)d_in[14];
    const float* b2  = (const float*)d_in[15];
    const int* src = (const int*)d_in[16];
    const int* dst = (const int*)d_in[17];
    float* out = (float*)d_out;

    constexpr int NDIS = 5000, NGEN = 20000, NN = 25000, E = 400000, EMB = 128, HID = 64;

    char* ws = (char*)d_ws;
    float* h    = (float*)(ws);                 // 25000*128 f32 = 12.8 MB
    float* agg  = (float*)(ws + 12800000);      // 25000*128 f32 = 12.8 MB (L2-resident)
    float* rdeg = (float*)(ws + 25600000);      // 25000 f32 (deg -> reciprocal in place)
    float* h1   = (float*)(ws + 25700096);      // 25000*64 f32 = 6.4 MB

    dim3 blk(256);
    auto gemmGrid = [](int M, int N) { return dim3((unsigned)((M + BM - 1) / BM), (unsigned)(N / BN)); };

    // ---- node embeddings: h = [0.9*(Xd@Wd+bd)+0.1*(miT@W1) ; 0.9*(Xg@Wg+bg)+0.1*(gd@W4)]
    gemm_wmma_f16<false, false><<<gemmGrid(NDIS, EMB), blk, 0, stream>>>(
        d_features, Wd, bd, nullptr, h, NDIS, EMB, 383, 383, 0.9f, 0, 0);
    gemm_wmma_f16<true, false><<<gemmGrid(NDIS, EMB), blk, 0, stream>>>(
        miRNA_dis, W1, nullptr, nullptr, h, NDIS, EMB, 2000, NDIS, 0.1f, 1, 0);
    gemm_wmma_f16<false, false><<<gemmGrid(NGEN, EMB), blk, 0, stream>>>(
        g_features, Wg, bg, nullptr, h + (long)NDIS * EMB, NGEN, EMB, 4395, 4395, 0.9f, 0, 0);
    gemm_wmma_f16<false, false><<<gemmGrid(NGEN, EMB), blk, 0, stream>>>(
        gene_dis, W4, nullptr, nullptr, h + (long)NDIS * EMB, NGEN, EMB, NDIS, NDIS, 0.1f, 1, 0);

    // ---- degree (shared by both layers, then inverted) + layer-1 aggregation
    fill_zero<<<(unsigned)(((long)NN * EMB + 255) / 256), blk, 0, stream>>>(agg, (long)NN * EMB);
    fill_zero<<<(NN + 255) / 256, blk, 0, stream>>>(rdeg, NN);
    deg_kernel<<<(E + 255) / 256, blk, 0, stream>>>(dst, rdeg, E);
    finalize_rdeg<<<(NN + 255) / 256, blk, 0, stream>>>(rdeg, NN);
    scatter_kernel<<<(unsigned)(((long)E * (EMB / 4) + 255) / 256), blk, 0, stream>>>(h, src, dst, agg, E, EMB);

    // ---- SAGE layer 1: h1 = relu( h@Ws1 + (agg*rdeg)@Wn1 + b1 )
    gemm_wmma_f16<false, false><<<gemmGrid(NN, HID), blk, 0, stream>>>(
        h, Ws1, b1, nullptr, h1, NN, HID, EMB, EMB, 1.0f, 0, 0);
    gemm_wmma_f16<false, true><<<gemmGrid(NN, HID), blk, 0, stream>>>(
        agg, Wn1, nullptr, rdeg, h1, NN, HID, EMB, EMB, 1.0f, 1, 1);

    // ---- layer-2 aggregation on h1
    fill_zero<<<(unsigned)(((long)NN * HID + 255) / 256), blk, 0, stream>>>(agg, (long)NN * HID);
    scatter_kernel<<<(unsigned)(((long)E * (HID / 4) + 255) / 256), blk, 0, stream>>>(h1, src, dst, agg, E, HID);

    // ---- SAGE layer 2: out = h1@Ws2 + (agg*rdeg)@Wn2 + b2
    gemm_wmma_f16<false, false><<<gemmGrid(NN, HID), blk, 0, stream>>>(
        h1, Ws2, b2, nullptr, out, NN, HID, HID, HID, 1.0f, 0, 0);
    gemm_wmma_f16<false, true><<<gemmGrid(NN, HID), blk, 0, stream>>>(
        agg, Wn2, nullptr, rdeg, out, NN, HID, HID, HID, 1.0f, 1, 0);
}
